// _GraphEncoderStack_57114475102424
// MI455X (gfx1250) — matrix operand order repaired and checked by reading, not compiled, not run on hardware
//
#include <hip/hip_runtime.h>
#include <hip/hip_bf16.h>
#include <math.h>

#define N_NODES 50000
#define N_EDGES 800000
#define IN_DIM  128
#define HIDDEN  128
#define LATENT  32
#define BN_EPS  1e-5f

typedef __attribute__((ext_vector_type(16))) __bf16 v16bf;
typedef __attribute__((ext_vector_type(8)))  __bf16 v8bf;
typedef __attribute__((ext_vector_type(8)))  float  v8f;

// ---------------- GCN normalization ----------------

__global__ void init_deg_kernel(float* __restrict__ deg, int n) {
    int i = blockIdx.x * blockDim.x + threadIdx.x;
    if (i < n) deg[i] = 1.0f;   // self-loop weight
}

__global__ void deg_accum_kernel(const int* __restrict__ dst, const float* __restrict__ w,
                                 float* __restrict__ deg, int ne) {
    int e = blockIdx.x * blockDim.x + threadIdx.x;
    if (e < ne) atomicAdd(&deg[dst[e]], w[e]);
}

__global__ void dinv_kernel(const float* __restrict__ deg, float* __restrict__ dinv, int n) {
    int i = blockIdx.x * blockDim.x + threadIdx.x;
    if (i < n) {
        float d = deg[i];
        dinv[i] = (d > 0.0f) ? rsqrtf(d) : 0.0f;
    }
}

__global__ void norm_kernel(const int* __restrict__ src, const int* __restrict__ dst,
                            const float* __restrict__ w, const float* __restrict__ dinv,
                            float* __restrict__ nrm, int ne) {
    int e = blockIdx.x * blockDim.x + threadIdx.x;
    if (e < ne) nrm[e] = dinv[src[e]] * w[e] * dinv[dst[e]];
}

// ---------------- bf16 pre-encode helpers ----------------

__global__ void f32_to_bf16_kernel(const float* __restrict__ src, __bf16* __restrict__ dst,
                                   long long count) {
    long long i = (long long)blockIdx.x * blockDim.x + threadIdx.x;
    if (i < count) dst[i] = (__bf16)src[i];
}

// W [K][OUT] f32 row-major  ->  Wt [OUT][K] bf16 (i.e., column-major of W)
__global__ void w_transpose_bf16_kernel(const float* __restrict__ W, __bf16* __restrict__ Wt,
                                        int K, int OUT) {
    int idx = blockIdx.x * blockDim.x + threadIdx.x;
    if (idx >= K * OUT) return;
    int n = idx / K;
    int k = idx % K;
    Wt[idx] = (__bf16)W[(size_t)k * OUT + n];
}

// ---------------- Dense GEMM via bf16 WMMA ----------------
// One wave = one 16x16 output tile. A: bf16 [n][K] row-major. B: bf16 [OUT][K]
// (column-major of the logical [K][OUT] weight). K,OUT compile-time -> full unroll.
// Per lane per WMMA: A = 2x contiguous b128, B = 1x contiguous 32B load.
template <int K, int OUT>
__global__ void __launch_bounds__(32)
gemm_wmma_kernel(const __bf16* __restrict__ Xb, const __bf16* __restrict__ Wt,
                 float* __restrict__ O) {
    const int lane = threadIdx.x;       // wave32
    const int m    = lane & 15;
    const int hf   = lane >> 4;         // lane half 0/1
    // A lane base: row = blockTile*16 + m, K-base = hf*16 (contiguous, matches B)
    const __bf16* arow = Xb + (size_t)(blockIdx.x * 16 + m) * K + hf * 16;
    // B lane base: col = colTile*16 + m, K-base = hf*16 (contiguous, matches A)
    const __bf16* brow = Wt + (size_t)(blockIdx.y * 16 + m) * K + hf * 16;

    v8f acc = {};
#pragma unroll
    for (int kk = 0; kk < K; kk += 32) {
        v16bf a = *(const v16bf*)(arow + kk);        // K = kbA + 0..15
        v16bf b = *(const v16bf*)(brow + kk);        // K = kbB + 0..15
        acc = __builtin_amdgcn_wmma_f32_16x16x32_bf16(
            /*neg_a=*/false, a, /*neg_b=*/false, b,
            /*c_mod=*/(short)0, acc, /*reuse_a=*/false, /*reuse_b=*/false);
    }
    // C/D layout: VGPR r -> M = r + hf*8, N = lane&15
    float* orow = O + (size_t)(blockIdx.x * 16 + hf * 8) * OUT + blockIdx.y * 16 + m;
#pragma unroll
    for (int r = 0; r < 8; ++r)
        orow[(size_t)r * OUT] = acc[r];
}

// ---------------- Message passing ----------------

// agg[i][f] = h[i][f]*dinv[i]^2 + bias[f]   (self-loop contribution + bias)
__global__ void self_bias_kernel(const float* __restrict__ h, const float* __restrict__ dinv,
                                 const float* __restrict__ bias, float* __restrict__ agg,
                                 int n, int F) {
    long long i = (long long)blockIdx.x * blockDim.x + threadIdx.x;
    if (i >= (long long)n * F) return;
    int node = (int)(i / F);
    int f    = (int)(i % F);
    float di = dinv[node];
    agg[i] = h[i] * di * di + bias[f];
}

// agg[dst] += h[src] * norm, 4 features per thread (float4 gather, f32 atomics in L2)
__global__ void edge_scatter_kernel(const float* __restrict__ h, const int* __restrict__ src,
                                    const int* __restrict__ dst, const float* __restrict__ nrm,
                                    float* __restrict__ agg, int ne, int F) {
    const int per = F >> 2;
    long long idx = (long long)blockIdx.x * blockDim.x + threadIdx.x;
    long long e = idx / per;
    if (e >= ne) return;
    int fo = ((int)(idx % per)) << 2;
    int s = src[e], d = dst[e];
    float w = nrm[e];
    const float4 hv = *(const float4*)(h + (size_t)s * F + fo);
    float* ap = agg + (size_t)d * F + fo;
    atomicAdd(ap + 0, hv.x * w);
    atomicAdd(ap + 1, hv.y * w);
    atomicAdd(ap + 2, hv.z * w);
    atomicAdd(ap + 3, hv.w * w);
}

// ---------------- BatchNorm ----------------

__global__ void __launch_bounds__(256)
bn_stats_kernel(const float* __restrict__ x, float* __restrict__ mean,
                float* __restrict__ rstd, int n, int F) {
    __shared__ float ssum[256];
    __shared__ float ssq[256];
    int f = blockIdx.x;
    float s = 0.0f, q = 0.0f;
    for (int i = threadIdx.x; i < n; i += 256) {
        float v = x[(size_t)i * F + f];
        s += v; q += v * v;
    }
    ssum[threadIdx.x] = s;
    ssq[threadIdx.x]  = q;
    __syncthreads();
    for (int off = 128; off > 0; off >>= 1) {
        if (threadIdx.x < off) {
            ssum[threadIdx.x] += ssum[threadIdx.x + off];
            ssq[threadIdx.x]  += ssq[threadIdx.x + off];
        }
        __syncthreads();
    }
    if (threadIdx.x == 0) {
        float mu  = ssum[0] / (float)n;
        float var = ssq[0] / (float)n - mu * mu;
        mean[f] = mu;
        rstd[f] = rsqrtf(var + BN_EPS);
    }
}

// out = relu((agg-mean)*rstd*g + be) [+ residual]; also emits bf16 copy for next GEMM
__global__ void bn_relu_kernel(const float* __restrict__ agg, const float* __restrict__ mean,
                               const float* __restrict__ rstd, const float* __restrict__ g,
                               const float* __restrict__ be, const float* __restrict__ residual,
                               float* __restrict__ out, __bf16* __restrict__ out_bf,
                               int n, int F) {
    long long i = (long long)blockIdx.x * blockDim.x + threadIdx.x;
    if (i >= (long long)n * F) return;
    int f = (int)(i % F);
    float v = (agg[i] - mean[f]) * rstd[f] * g[f] + be[f];
    v = fmaxf(v, 0.0f);
    if (residual) v += residual[i];
    out[i] = v;
    out_bf[i] = (__bf16)v;
}

// ---------------- Reparameterization / output fan-out ----------------

__global__ void reparam_kernel(const float* __restrict__ qm, const float* __restrict__ qs,
                               const float* __restrict__ eps, float* __restrict__ out,
                               int n, int L) {
    long long i = (long long)blockIdx.x * blockDim.x + threadIdx.x;
    long long NL = (long long)n * L;
    if (i >= NL) return;
    float m = qm[i], s = qs[i];
    float sp = (s > 20.0f) ? s : log1pf(expf(s));   // softplus
    float sd = sp + 1e-6f;
    out[i]          = m + sd * eps[i];  // q_z
    out[NL + i]     = m;                // q_m
    out[2 * NL + i] = s;                // q_s
}

// ---------------- Launcher ----------------

extern "C" void kernel_launch(void* const* d_in, const int* in_sizes, int n_in,
                              void* d_out, int out_size, void* d_ws, size_t ws_size,
                              hipStream_t stream) {
    (void)in_sizes; (void)n_in; (void)out_size; (void)ws_size;

    const float* x   = (const float*)d_in[0];
    const int*   ei  = (const int*)  d_in[1];
    const float* ew  = (const float*)d_in[2];
    const float* W0  = (const float*)d_in[3];
    const float* b0  = (const float*)d_in[4];
    const float* W1  = (const float*)d_in[5];
    const float* b1  = (const float*)d_in[6];
    const float* g0  = (const float*)d_in[7];
    const float* be0 = (const float*)d_in[8];
    const float* g1  = (const float*)d_in[9];
    const float* be1 = (const float*)d_in[10];
    const float* Wm  = (const float*)d_in[11];
    const float* bm  = (const float*)d_in[12];
    const float* Wsp = (const float*)d_in[13];
    const float* bsp = (const float*)d_in[14];
    const float* eps = (const float*)d_in[15];
    const int* src = ei;              // edge_index row 0
    const int* dst = ei + N_EDGES;    // edge_index row 1
    float* out = (float*)d_out;

    // carve scratch
    char* ws = (char*)d_ws;
    size_t off = 0;
    auto alloc = [&](size_t bytes) {
        void* p = (void*)(ws + off);
        off += ((bytes + 255) / 256) * 256;
        return p;
    };
    float*  deg  = (float*) alloc(N_NODES * 4);
    float*  dinv = (float*) alloc(N_NODES * 4);
    float*  nrm  = (float*) alloc((size_t)N_EDGES * 4);
    float*  h    = (float*) alloc((size_t)N_NODES * HIDDEN * 4);
    float*  agg  = (float*) alloc((size_t)N_NODES * HIDDEN * 4);
    float*  res  = (float*) alloc((size_t)N_NODES * HIDDEN * 4);
    float*  hm   = (float*) alloc((size_t)N_NODES * LATENT * 4);
    float*  hs   = (float*) alloc((size_t)N_NODES * LATENT * 4);
    float*  am   = (float*) alloc((size_t)N_NODES * LATENT * 4);
    float*  as   = (float*) alloc((size_t)N_NODES * LATENT * 4);
    float*  mean = (float*) alloc(HIDDEN * 4);
    float*  rstd = (float*) alloc(HIDDEN * 4);
    __bf16* xb   = (__bf16*)alloc((size_t)N_NODES * HIDDEN * 2);  // bf16 activations
    __bf16* resb = (__bf16*)alloc((size_t)N_NODES * HIDDEN * 2);
    __bf16* W0t  = (__bf16*)alloc((size_t)HIDDEN * IN_DIM * 2);   // [OUT][K] bf16
    __bf16* W1t  = (__bf16*)alloc((size_t)HIDDEN * HIDDEN * 2);
    __bf16* Wmt  = (__bf16*)alloc((size_t)LATENT * HIDDEN * 2);
    __bf16* Wst  = (__bf16*)alloc((size_t)LATENT * HIDDEN * 2);

    const int TB = 256;
    const long long nh  = (long long)N_NODES * HIDDEN;
    const long long nl  = (long long)N_NODES * LATENT;
    const long long esH = (long long)N_EDGES * (HIDDEN / 4);
    const long long esL = (long long)N_EDGES * (LATENT / 4);
    const int gNH = (int)((nh + TB - 1) / TB);
    const int gNL = (int)((nl + TB - 1) / TB);

    // --- normalization coefficients ---
    init_deg_kernel<<<(N_NODES + TB - 1) / TB, TB, 0, stream>>>(deg, N_NODES);
    deg_accum_kernel<<<(N_EDGES + TB - 1) / TB, TB, 0, stream>>>(dst, ew, deg, N_EDGES);
    dinv_kernel<<<(N_NODES + TB - 1) / TB, TB, 0, stream>>>(deg, dinv, N_NODES);
    norm_kernel<<<(N_EDGES + TB - 1) / TB, TB, 0, stream>>>(src, dst, ew, dinv, nrm, N_EDGES);

    // --- one-time bf16 encodes: input features + all weight transposes ---
    f32_to_bf16_kernel<<<gNH, TB, 0, stream>>>(x, xb, nh);
    w_transpose_bf16_kernel<<<(IN_DIM * HIDDEN + TB - 1) / TB, TB, 0, stream>>>(W0, W0t, IN_DIM, HIDDEN);
    w_transpose_bf16_kernel<<<(HIDDEN * HIDDEN + TB - 1) / TB, TB, 0, stream>>>(W1, W1t, HIDDEN, HIDDEN);
    w_transpose_bf16_kernel<<<(HIDDEN * LATENT + TB - 1) / TB, TB, 0, stream>>>(Wm, Wmt, HIDDEN, LATENT);
    w_transpose_bf16_kernel<<<(HIDDEN * LATENT + TB - 1) / TB, TB, 0, stream>>>(Wsp, Wst, HIDDEN, LATENT);

    dim3 gH(N_NODES / 16, HIDDEN / 16);
    dim3 gL(N_NODES / 16, LATENT / 16);

    // --- layer 0: GEMM -> aggregate -> BN -> ReLU (residual saved in res/resb) ---
    gemm_wmma_kernel<IN_DIM, HIDDEN><<<gH, 32, 0, stream>>>(xb, W0t, h);
    self_bias_kernel<<<gNH, TB, 0, stream>>>(h, dinv, b0, agg, N_NODES, HIDDEN);
    edge_scatter_kernel<<<(int)((esH + TB - 1) / TB), TB, 0, stream>>>(h, src, dst, nrm, agg, N_EDGES, HIDDEN);
    bn_stats_kernel<<<HIDDEN, 256, 0, stream>>>(agg, mean, rstd, N_NODES, HIDDEN);
    bn_relu_kernel<<<gNH, TB, 0, stream>>>(agg, mean, rstd, g0, be0, nullptr, res, resb, N_NODES, HIDDEN);

    // --- layer 1: GEMM -> aggregate -> BN -> ReLU -> + residual (in place on res) ---
    gemm_wmma_kernel<HIDDEN, HIDDEN><<<gH, 32, 0, stream>>>(resb, W1t, h);
    self_bias_kernel<<<gNH, TB, 0, stream>>>(h, dinv, b1, agg, N_NODES, HIDDEN);
    edge_scatter_kernel<<<(int)((esH + TB - 1) / TB), TB, 0, stream>>>(h, src, dst, nrm, agg, N_EDGES, HIDDEN);
    bn_stats_kernel<<<HIDDEN, 256, 0, stream>>>(agg, mean, rstd, N_NODES, HIDDEN);
    bn_relu_kernel<<<gNH, TB, 0, stream>>>(agg, mean, rstd, g1, be1, res, res, resb, N_NODES, HIDDEN);

    // --- heads: q_m and q_s GCNs ---
    gemm_wmma_kernel<HIDDEN, LATENT><<<gL, 32, 0, stream>>>(resb, Wmt, hm);
    gemm_wmma_kernel<HIDDEN, LATENT><<<gL, 32, 0, stream>>>(resb, Wst, hs);
    self_bias_kernel<<<gNL, TB, 0, stream>>>(hm, dinv, bm, am, N_NODES, LATENT);
    self_bias_kernel<<<gNL, TB, 0, stream>>>(hs, dinv, bsp, as, N_NODES, LATENT);
    edge_scatter_kernel<<<(int)((esL + TB - 1) / TB), TB, 0, stream>>>(hm, src, dst, nrm, am, N_EDGES, LATENT);
    edge_scatter_kernel<<<(int)((esL + TB - 1) / TB), TB, 0, stream>>>(hs, src, dst, nrm, as, N_EDGES, LATENT);

    // --- reparameterize + write (q_z, q_m, q_s) ---
    reparam_kernel<<<gNL, TB, 0, stream>>>(am, as, eps, out, N_NODES, LATENT);
}
